// seq2seq_7524782702891
// MI455X (gfx1250) — matrix operand order repaired and checked by reading, not compiled, not run on hardware
//
#include <hip/hip_runtime.h>
#include <hip/hip_bf16.h>

// MI455X (gfx1250) seq2seq decode — persistent kernel, 128 WGs x 256 threads.
// Residency plan (verified by codegen in prior rounds):
//   - dec_Wih gate tiles: 24 fragments = 192 VGPRs/wave, loaded once.
//   - dec_Whh gate tiles: 192KB dynamic LDS per WG, copied once.
//   - out_W column block (32KB), biases, out_b: LDS/registers, one-time.
// Per step: ds_load_b128 + 48 v_wmma_f32_16x16x32_bf16 per wave, bf16 h
// ping-pong via L2, ONE sense-reversing grid barrier.
// This round: barrier poll is a relaxed load (not an RMW) so spinners don't
// serialize in the L2 atomic unit, and the logits scatter is split into 4
// slices (fan-in per address 128 -> 32); the redundant per-WG log-softmax
// reduction sums the slices on load.

#define HDIM 2048
#define ODIM 1024
#define NWG  128
#define TPB  256
#define LEAKK 0.01f
#define LSLICE 4                      // logits fan-in slices

typedef __attribute__((ext_vector_type(16))) __bf16 v16bf;
typedef __attribute__((ext_vector_type(8)))  float  v8f;

struct Frag32 { uint4 lo; uint4 hi; };   // 32 bytes == one bf16 fragment per lane

__device__ __forceinline__ unsigned short f2bf(float f) {
  unsigned u = __float_as_uint(f);
  unsigned r = u + 0x7FFFu + ((u >> 16) & 1u);   // round-to-nearest-even
  return (unsigned short)(r >> 16);
}
__device__ __forceinline__ float bf2f(unsigned short b) {
  return __uint_as_float(((unsigned)b) << 16);
}
__device__ __forceinline__ float fast_sigmoid(float x) {
  return __builtin_amdgcn_rcpf(1.0f + __expf(-x));   // rcp(inf)=0 -> safe
}
__device__ __forceinline__ float fast_tanh(float x) {
  x = fminf(fmaxf(x, -15.0f), 15.0f);                // keep exp finite
  float e2 = __expf(2.0f * x);
  return (e2 - 1.0f) * __builtin_amdgcn_rcpf(e2 + 1.0f);
}

// ---------------- one-time packing kernels ----------------

// Pack a (6144 x 2048) fp32 matrix into bf16 WMMA A-fragment tiles.
// Tile (tm, kc) = rows [tm*16,+16) x K [kc*32,+32), 1024 bytes:
//   half h in {0,1}, lane L in [0,32), dword dw in [0,4):
//   M = tm*16 + (L & 15);  K = kc*32 + (L>=16 ? 8 : 0) + h*16 + dw*2 (+0,+1)
__global__ void pack_tiles_k(const float* __restrict__ W, unsigned* __restrict__ dst) {
  const int b    = blockIdx.x;            // tm*64 + kc
  const int tm   = b >> 6, kc = b & 63;
  const int o    = threadIdx.x << 2;      // byte offset within 1 KB tile
  const int half = o >> 9, lane = (o >> 4) & 31, dw = (o >> 2) & 3;
  const int m = tm * 16 + (lane & 15);
  const int k = kc * 32 + ((lane >> 4) * 8) + half * 16 + dw * 2;
  const float f0 = W[(size_t)m * HDIM + k];
  const float f1 = W[(size_t)m * HDIM + k + 1];
  dst[(size_t)b * 256 + threadIdx.x] = (unsigned)f2bf(f0) | ((unsigned)f2bf(f1) << 16);
}

__global__ void pack_rm_k(const float* __restrict__ W, unsigned* __restrict__ dst, int nPairs) {
  int i = blockIdx.x * blockDim.x + threadIdx.x;
  if (i < nPairs)
    dst[i] = (unsigned)f2bf(W[2 * i]) | ((unsigned)f2bf(W[2 * i + 1]) << 16);
}

__global__ void init_bar_k(unsigned* bar) { bar[0] = 0u; bar[1] = 0u; }

// ---------------- persistent kernel helpers ----------------

__device__ __forceinline__ void gridBarrier(unsigned* bar) {
  __syncthreads();
  if (threadIdx.x == 0) {
    __threadfence();
    unsigned g = __hip_atomic_load(&bar[1], __ATOMIC_RELAXED,
                                   __HIP_MEMORY_SCOPE_AGENT);   // generation first
    unsigned arrived = atomicAdd(&bar[0], 1u);
    if (arrived == (unsigned)(NWG - 1)) {
      atomicExch(&bar[0], 0u);
      __threadfence();
      atomicAdd(&bar[1], 1u);                     // release
    } else {
      // plain-load poll: concurrent L2 service, no RMW serialization
      while (__hip_atomic_load(&bar[1], __ATOMIC_RELAXED,
                               __HIP_MEMORY_SCOPE_AGENT) == g)
        __builtin_amdgcn_s_sleep(2);
    }
    __threadfence();                              // acquire
  }
  __syncthreads();
}

// topv = max(log_softmax(l)) = -log(sum_i exp(l_i - max l)) over 1024 logits,
// where logits are spread over LSLICE partial slices of 1024 each.
__device__ float reduce_topv(const float* __restrict__ l, float* sRed) {
  const int tid = threadIdx.x;
  float v0 = l[tid]       + l[1024 + tid]       + l[2048 + tid]       + l[3072 + tid];
  float v1 = l[tid + 256] + l[1024 + tid + 256] + l[2048 + tid + 256] + l[3072 + tid + 256];
  float v2 = l[tid + 512] + l[1024 + tid + 512] + l[2048 + tid + 512] + l[3072 + tid + 512];
  float v3 = l[tid + 768] + l[1024 + tid + 768] + l[2048 + tid + 768] + l[3072 + tid + 768];
  float m = fmaxf(fmaxf(v0, v1), fmaxf(v2, v3));
  sRed[tid] = m; __syncthreads();
  for (int s = TPB / 2; s > 0; s >>= 1) {
    if (tid < s) sRed[tid] = fmaxf(sRed[tid], sRed[tid + s]);
    __syncthreads();
  }
  const float M = sRed[0]; __syncthreads();
  float su = __expf(v0 - M) + __expf(v1 - M) + __expf(v2 - M) + __expf(v3 - M);
  sRed[tid] = su; __syncthreads();
  for (int s = TPB / 2; s > 0; s >>= 1) {
    if (tid < s) sRed[tid] += sRed[tid + s];
    __syncthreads();
  }
  const float S = sRed[0]; __syncthreads();
  return -__logf(S);
}

// ---------------- persistent kernel ----------------

__global__ void __launch_bounds__(TPB, 1) seq2seq_persist(
    const float* __restrict__ a,
    const float* __restrict__ encWih,
    const float* __restrict__ enc_bih,
    const float* __restrict__ enc_bhh,
    const float* __restrict__ intoW,
    const float* __restrict__ intoB,
    const float* __restrict__ dec_bih,
    const float* __restrict__ dec_bhh,
    const float* __restrict__ out_b,
    const unsigned* __restrict__ tIh,   // packed dec_Wih tiles (bf16 frags)
    const unsigned* __restrict__ tHh,   // packed dec_Whh tiles (bf16 frags)
    const unsigned* __restrict__ owbf,  // out_W bf16 row-major (O x H), uint pairs
    unsigned* __restrict__ hbf,         // 2 x 1024 uints: h ping-pong, bf16 pairs
    float* __restrict__ lbuf,           // 3 x LSLICE x O triple-buffered logits
    unsigned* __restrict__ bar,
    float* __restrict__ out)
{
  // dynamic LDS: 3 Whh tiles (49152 uints) + out_W column block (8192 uints)
  extern __shared__ unsigned dynLds[];
  unsigned* sTiles = dynLds;              // 192 KB
  unsigned* sOW    = dynLds + 3 * 16384;  // 32 KB
  __shared__ float    sIW[HDIM];          // into_W
  __shared__ float    sIB[HDIM];          // into_b
  __shared__ unsigned sVbf[HDIM / 2];     // v as packed bf16 pairs
  __shared__ unsigned sHbf[HDIM / 2];     // h_prev as packed bf16 pairs
  __shared__ float    sBias[96];          // gate biases for this WG's 16 rows
  __shared__ float    sPart[96];          // 6 gates x 16 rows partial sums
  __shared__ float    sHnew[16];          // own rows' h (f32), persists across steps
  __shared__ float    sRed[TPB];

  const int tid  = threadIdx.x;
  const int wg   = blockIdx.x;
  const int m0   = wg * 16;               // this WG's 16 h-rows
  const int wave = tid >> 5;
  const int lane = tid & 31;

  // ---- one-time residency fills ----
  for (int j = tid; j < HDIM; j += TPB) { sIW[j] = intoW[j]; sIB[j] = intoB[j]; }
  for (int idx = tid; idx < 3 * 16384; idx += TPB) {      // Whh tiles -> LDS
    const int q = idx >> 14, off = idx & 16383;
    sTiles[idx] = tHh[(((size_t)(q * NWG + wg) * 64) << 8) + off];
  }
  for (int idx = tid; idx < 8192; idx += TPB) {           // out_W cols -> LDS
    const int i = idx >> 3, jj = idx & 7;
    sOW[idx] = owbf[(size_t)i * (HDIM / 2) + (m0 >> 1) + jj];
  }
  if (tid < 96) {                                         // gate biases
    const int g = tid >> 4, r = tid & 15;
    sBias[tid] = (g < 3) ? dec_bih[g * HDIM + m0 + r]
                         : dec_bhh[(g - 3) * HDIM + m0 + r];
  }
  const float myOutB = (tid < 32) ? out_b[wg * 8 + (tid & 7)] : 0.0f;

  // Wih tiles -> registers: 3 gates x this wave's 8 K-chunks = 192 VGPRs
  const int kc0   = wave * 8;
  const int laneB = lane << 2;                            // uint offset of A frag
  v16bf Areg[24];
  #pragma unroll
  for (int q = 0; q < 3; ++q)
    #pragma unroll
    for (int c = 0; c < 8; ++c) {
      const unsigned* p = tIh + (((size_t)(q * NWG + wg) * 64 + kc0 + c) << 8) + laneB;
      Frag32 fa;
      fa.lo = *(const uint4*)p;
      fa.hi = *(const uint4*)(p + 128);
      Areg[q * 8 + c] = __builtin_bit_cast(v16bf, fa);
    }

  // ---- encoder (one-time): h0 = GRUCell(a, 0); Whh@0 == 0 so gh == enc_bhh ----
  if (tid < 48) {
    const int g = tid >> 4, r = tid & 15;
    const int row = g * HDIM + m0 + r;
    const float* wr = encWih + (size_t)row * HDIM;
    float s = enc_bih[row];
    for (int k = 0; k < HDIM; ++k) s += wr[k] * a[k];
    sPart[tid] = s;
  }
  __syncthreads();
  if (tid < 16) {
    const int m = m0 + tid;
    float rr = fast_sigmoid(sPart[tid]      + enc_bhh[m]);
    float zz = fast_sigmoid(sPart[16 + tid] + enc_bhh[HDIM + m]);
    float nn = fast_tanh(sPart[32 + tid] + rr * enc_bhh[2 * HDIM + m]);
    float h0 = (1.0f - zz) * nn;
    sHnew[tid] = h0;                                      // own rows' f32 h
    ((unsigned short*)hbf)[2048 + m] = f2bf(h0);          // hbf[1], step 0 reads prev=1
  }
  if (tid < 32) {                                         // seed lbuf[0] slices
    const int s = tid >> 3;
    lbuf[s * ODIM + wg * 8 + (tid & 7)] = (s == 0) ? myOutB : 0.0f;
  }
  __threadfence();
  gridBarrier(bar);

  const int bOff  = kc0 * 16 + ((lane >> 4) << 2);        // uint offset of B fragment
  const int aBase = (kc0 << 8) + laneB;                   // LDS A frag base (uints)

  // ---- 1024 decode steps, one grid barrier each ----
  for (int t = 0; t < ODIM; ++t) {
    float topv;
    if (t > 0) {
      topv = reduce_topv(lbuf + ((t - 1) % 3) * (LSLICE * ODIM), sRed);
      if (wg == 0 && tid == 0) out[t - 1] = topv;
    } else {
      topv = 0.0f;                                             // initial decoder input
    }
    __syncthreads();

    const int prev = (t + 1) & 1, cur = t & 1;
    // v = LeakyReLU(into_W * topv + into_b) -> bf16 LDS; h_prev bf16 straight copy
    const unsigned* hprevU = hbf + prev * (HDIM / 2);
    for (int p = tid; p < HDIM / 2; p += TPB) {
      float va = fmaf(sIW[2 * p],     topv, sIB[2 * p]);
      float vb = fmaf(sIW[2 * p + 1], topv, sIB[2 * p + 1]);
      va = va > 0.0f ? va : LEAKK * va;
      vb = vb > 0.0f ? vb : LEAKK * vb;
      sVbf[p] = (unsigned)f2bf(va) | ((unsigned)f2bf(vb) << 16);
      sHbf[p] = hprevU[p];
    }
    if (tid < 32) {                                            // pre-seed next logits buf
      const int s = tid >> 3;
      lbuf[((t + 1) % 3) * (LSLICE * ODIM) + s * ODIM + wg * 8 + (tid & 7)] =
          (s == 0) ? myOutB : 0.0f;
    }
    if (tid < 96) sPart[tid] = sBias[tid];                     // gate bias init
    __syncthreads();

    // 6 accumulation chains: gates 0-2 A from registers (B=v),
    // gates 3-5 A from LDS (B=h). 48 WMMAs per wave per step.
    v8f acc[6];
    #pragma unroll
    for (int g = 0; g < 6; ++g) acc[g] = (v8f){0,0,0,0,0,0,0,0};
    #pragma unroll
    for (int c = 0; c < 8; ++c) {
      const unsigned* vb = sVbf + bOff + c * 16;
      const unsigned* hb = sHbf + bOff + c * 16;
      Frag32 bv, bh;
      bv.lo = *(const uint4*)vb;  bv.hi = *(const uint4*)(vb + 8);
      bh.lo = *(const uint4*)hb;  bh.hi = *(const uint4*)(hb + 8);
      const v16bf Bv = __builtin_bit_cast(v16bf, bv);
      const v16bf Bh = __builtin_bit_cast(v16bf, bh);
      #pragma unroll
      for (int q = 0; q < 3; ++q)
        acc[q] = __builtin_amdgcn_wmma_f32_16x16x32_bf16(
            false, Areg[q * 8 + c], false, Bv, (short)0, acc[q], false, false);
      #pragma unroll
      for (int q = 0; q < 3; ++q) {
        const unsigned* ap = sTiles + (q << 14) + aBase + (c << 8);
        Frag32 fa;
        fa.lo = *(const uint4*)ap;                 // ds_load_b128
        fa.hi = *(const uint4*)(ap + 128);
        const v16bf A = __builtin_bit_cast(v16bf, fa);
        acc[3 + q] = __builtin_amdgcn_wmma_f32_16x16x32_bf16(
            false, A, false, Bh, (short)0, acc[3 + q], false, false);
      }
    }
    // D column 0: lane 0 holds rows m0..m0+7, lane 16 holds m0+8..m0+15
    if ((lane & 15) == 0) {
      const int r0 = (lane >> 4) * 8;
      #pragma unroll
      for (int g = 0; g < 6; ++g)
        #pragma unroll
        for (int r = 0; r < 8; ++r)
          atomicAdd(&sPart[g * 16 + r0 + r], acc[g][r]);
    }
    __syncthreads();

    if (tid < 16) {                                            // GRU combine, 16 rows
      float p0 = sPart[tid],      p1 = sPart[16 + tid], p2 = sPart[32 + tid];
      float p3 = sPart[48 + tid], p4 = sPart[64 + tid], p5 = sPart[80 + tid];
      float rr = fast_sigmoid(p0 + p3);
      float zz = fast_sigmoid(p1 + p4);
      float nn = fast_tanh(p2 + rr * p5);
      float hn = (1.0f - zz) * nn + zz * sHnew[tid];           // own prev h (f32)
      sHnew[tid] = hn;
      ((unsigned short*)hbf)[cur * HDIM + m0 + tid] = f2bf(hn);
    }
    __syncthreads();

    // partial logits from LDS-resident out_W columns; scatter into this WG's
    // slice (fan-in per address: NWG/LSLICE = 32)
    float* ldst = lbuf + (t % 3) * (LSLICE * ODIM) + (wg & (LSLICE - 1)) * ODIM;
    #pragma unroll
    for (int ii = 0; ii < 4; ++ii) {
      const int i = tid + ii * TPB;
      const uint4* wp = (const uint4*)(sOW + i * 8);
      uint4 w0 = wp[0], w1 = wp[1];
      const unsigned wv[8] = {w0.x, w0.y, w0.z, w0.w, w1.x, w1.y, w1.z, w1.w};
      float s = 0.0f;
      #pragma unroll
      for (int j = 0; j < 8; ++j) {
        s += bf2f((unsigned short)(wv[j] & 0xFFFFu)) * sHnew[2 * j];
        s += bf2f((unsigned short)(wv[j] >> 16))     * sHnew[2 * j + 1];
      }
      atomicAdd(&ldst[i], s);
    }
    __threadfence();
    gridBarrier(bar);
  }

  // final topv from the last step's logits
  float tv = reduce_topv(lbuf + ((ODIM - 1) % 3) * (LSLICE * ODIM), sRed);
  if (wg == 0 && tid == 0) out[ODIM - 1] = tv;
}

// ---------------- launch ----------------

extern "C" void kernel_launch(void* const* d_in, const int* in_sizes, int n_in,
                              void* d_out, int out_size, void* d_ws, size_t ws_size,
                              hipStream_t stream) {
  (void)in_sizes; (void)n_in; (void)out_size; (void)ws_size;
  const float* a       = (const float*)d_in[0];
  const float* encWih  = (const float*)d_in[1];
  /* d_in[2] enc_Whh unused: h0 == 0 */
  const float* enc_bih = (const float*)d_in[3];
  const float* enc_bhh = (const float*)d_in[4];
  const float* intoW   = (const float*)d_in[5];
  const float* intoB   = (const float*)d_in[6];
  const float* decWih  = (const float*)d_in[7];
  const float* decWhh  = (const float*)d_in[8];
  const float* dec_bih = (const float*)d_in[9];
  const float* dec_bhh = (const float*)d_in[10];
  const float* outW    = (const float*)d_in[11];
  const float* out_b   = (const float*)d_in[12];
  float* out = (float*)d_out;

  unsigned char* ws = (unsigned char*)d_ws;
  const size_t tilesBytes = (size_t)384 * 64 * 1024;          // 25,165,824 per matrix
  unsigned* tIh  = (unsigned*)ws;
  unsigned* tHh  = (unsigned*)(ws + tilesBytes);
  unsigned* owbf = (unsigned*)(ws + 2 * tilesBytes);
  unsigned* hbf  = (unsigned*)(ws + 2 * tilesBytes + (size_t)ODIM * HDIM * 2);
  float*    lbuf = (float*)(hbf + 2 * (HDIM / 2));
  unsigned* bar  = (unsigned*)(lbuf + 3 * LSLICE * ODIM);
  // total ws use: ~54.6 MB

  pack_tiles_k<<<384 * 64, 256, 0, stream>>>(decWih, tIh);
  pack_tiles_k<<<384 * 64, 256, 0, stream>>>(decWhh, tHh);
  pack_rm_k<<<(ODIM * HDIM / 2 + 255) / 256, 256, 0, stream>>>(outW, owbf, ODIM * HDIM / 2);
  init_bar_k<<<1, 1, 0, stream>>>(bar);

  const size_t dynLdsBytes = (size_t)(3 * 16384 + 8192) * sizeof(unsigned); // 224 KB
  seq2seq_persist<<<NWG, TPB, dynLdsBytes, stream>>>(
      a, encWih, enc_bih, enc_bhh, intoW, intoB, dec_bih, dec_bhh, out_b,
      tIh, tHh, owbf, hbf, lbuf, bar, out);
}
